// Point_Transformer_12584254177585
// MI455X (gfx1250) — matrix operand order repaired and checked by reading, not compiled, not run on hardware
//
#include <hip/hip_runtime.h>
#include <hip/hip_bf16.h>

typedef _Float16 h16;
typedef __attribute__((ext_vector_type(16))) _Float16 v16h;
typedef __attribute__((ext_vector_type(8)))  _Float16 v8h;
typedef __attribute__((ext_vector_type(8)))  float    v8f;
typedef __attribute__((ext_vector_type(4)))  float    v4f;

#define Bb   8
#define Nn   8192
#define Kk   10
#define CH   64
#define TN   16
#define COLS (TN*Kk)   // 160 columns = 16 n * 10 k
#define ASTR 80        // padded channel stride in halves (160B: keeps v16h 32B-aligned)

// f32 -> v8h from two contiguous v4f global loads (32B aligned)
__device__ __forceinline__ v8h cvt8(const float* p) {
  v4f x = ((const v4f*)p)[0];
  v4f y = ((const v4f*)p)[1];
  v8h r;
  r[0]=(h16)x[0]; r[1]=(h16)x[1]; r[2]=(h16)x[2]; r[3]=(h16)x[3];
  r[4]=(h16)y[0]; r[5]=(h16)y[1]; r[6]=(h16)y[2]; r[7]=(h16)y[3];
  return r;
}

// A-operand: 16x32 f16 tile of a 64x64 row-major weight matrix.
// lane<16: K = ks+j (j<16) ; lane>=16: +16 on the K base.
__device__ __forceinline__ v16h loadA(const float* __restrict__ W, int mt, int ks, int lane) {
  int m  = mt*16 + (lane & 15);
  int kh = (lane >> 4) * 16;
  const float* p = W + m*64 + ks + kh;
  v8h lo = cvt8(p);
  v8h hi = cvt8(p + 8);
  return __builtin_shufflevector(lo, hi, 0,1,2,3,4,5,6,7,8,9,10,11,12,13,14,15);
}

// B-operand: 32x16 f16 tile from LDS activations stored [col][ch].
__device__ __forceinline__ v16h loadB(const h16* act, int col, int ks, int lane) {
  return *(const v16h*)(act + col*ASTR + ks + ((lane >> 4) << 4));
}

__device__ __forceinline__ v8f wmma_f16(v16h a, v16h b, v8f c) {
  return __builtin_amdgcn_wmma_f32_16x16x32_f16(false, a, false, b, (short)0, c, false, false);
}

__device__ __forceinline__ void store8h(h16* dst, v8f v) {
  v8h r;
  #pragma unroll
  for (int i = 0; i < 8; ++i) r[i] = (h16)v[i];
  *(v8h*)dst = r;
}

__global__ __launch_bounds__(256, 1)
void pt_fused_kernel(const float* __restrict__ feats, const float* __restrict__ cent,
                     const float* __restrict__ smp,
                     const float* __restrict__ W0,  const float* __restrict__ b0,
                     const float* __restrict__ W1,  const float* __restrict__ b1,
                     const float* __restrict__ W2,  const float* __restrict__ b2,
                     const float* __restrict__ W3,  const float* __restrict__ b3,
                     const float* __restrict__ Wp1, const float* __restrict__ Wp2,
                     const float* __restrict__ Ww1, const float* __restrict__ Ww2,
                     const float* __restrict__ bw2, const float* __restrict__ Wout,
                     const float* __restrict__ bout, float* __restrict__ out)
{
  __shared__ __align__(32) h16  sA [COLS*ASTR];   // feats f16, later pact
  __shared__ __align__(32) h16  sXN[COLS*ASTR];   // xn
  __shared__ __align__(32) h16  sX3[COLS*ASTR];   // x3 + ptf
  __shared__ __align__(32) h16  sXF[COLS*ASTR];   // xfs = x1 - x2 + ptf
  __shared__ __align__(32) h16  sW1t[8*Kk*CH];    // Ww1 transposed f16 [o8][k][m]
  __shared__ __align__(32) float sX1[CH*TN];      // x1 [o][n]
  __shared__ float sH [8*TN];                     // h  [o8][n]
  __shared__ float sWn[8*TN*Kk];                  // softmax weights [s][n][k]
  __shared__ __align__(32) h16  sO1[TN*ASTR];     // out1 [n][o]

  const int tid  = threadIdx.x;
  const int lane = tid & 31;
  const int wave = tid >> 5;
  const int b    = blockIdx.y;
  const int n0   = blockIdx.x * TN;

  // ---- Phase 1a: stage feats tile with register-blocked transpose ----
  // Blocks of 8 channels x 4 cols: 8 global b128 loads -> 4 v8h LDS stores.
  {
    #pragma unroll
    for (int rep = 0; rep < 2; ++rep) {
      int blk = tid + rep*256;
      if (blk < 320) {                      // 8 ch-octets * 40 col-quads
        int c0 = (blk & 7) * 8;
        int q4 = (blk >> 3) * 4;
        v4f rows[8];
        #pragma unroll
        for (int i = 0; i < 8; ++i)
          rows[i] = *(const v4f*)(feats + ((size_t)(b*CH + c0 + i)*Nn + n0)*Kk + q4);
        #pragma unroll
        for (int cc = 0; cc < 4; ++cc) {
          v8h p;
          #pragma unroll
          for (int i = 0; i < 8; ++i) p[i] = (h16)rows[i][cc];
          *(v8h*)&sA[(q4 + cc)*ASTR + c0] = p;
        }
      }
    }
  }
  // ---- Phase 1b: stage Ww1 transposed into LDS f16 [o8][k][m] ----
  {
    #pragma unroll
    for (int r = 0; r < 20; ++r) {
      int d = tid + r*256;                  // 5120 elements
      int o8 = d / (Kk*CH);
      int rem = d - o8*(Kk*CH);
      int k = rem >> 6, m = rem & 63;
      sW1t[d] = (h16)Ww1[o8*(CH*Kk) + m*Kk + k];
    }
  }
  __syncthreads();

  // ---- Phase 2: xn = relu(W0 @ feats + b0) -> sXN ----
  {
    const int mt = wave & 3;                // invariant per wave
    const v16h a0 = loadA(W0, mt,  0, lane);
    const v16h a1 = loadA(W0, mt, 32, lane);
    const int ob = mt*16 + ((lane >> 4) << 3);
    float bias[8];
    #pragma unroll
    for (int i = 0; i < 8; ++i) bias[i] = b0[ob+i];
    #pragma unroll
    for (int j = 0; j < 5; ++j) {
      int nt  = (wave + j*8) >> 2;
      int col = nt*16 + (lane & 15);
      v8f acc = {};
      acc = wmma_f16(a0, loadB(sA, col,  0, lane), acc);
      acc = wmma_f16(a1, loadB(sA, col, 32, lane), acc);
      v8f r;
      #pragma unroll
      for (int i = 0; i < 8; ++i) { float t = acc[i] + bias[i]; r[i] = t > 0.f ? t : 0.f; }
      store8h(&sXN[col*ASTR + ob], r);
    }
  }
  __syncthreads();

  // ---- Phase 3: pact = relu(Wp1 * ptsn) overwrites sA ----
  if (tid < COLS) {
    int col = tid;
    int n = col / Kk, k = col % Kk;
    size_t nn = (size_t)n0 + n;
    float pn = 0.f;
    #pragma unroll
    for (int d = 0; d < 3; ++d) {
      float diff = cent[((size_t)b*Nn + nn)*3 + d]
                 - smp[(((size_t)b*3 + d)*Nn + nn)*Kk + k];
      pn += diff * diff;
    }
    #pragma unroll
    for (int g = 0; g < 8; ++g) {
      v8h p;
      #pragma unroll
      for (int i = 0; i < 8; ++i) {
        float t = Wp1[g*8 + i] * pn;
        p[i] = (h16)(t > 0.f ? t : 0.f);
      }
      *(v8h*)&sA[col*ASTR + g*8] = p;
    }
  }
  __syncthreads();

  // ---- Phase 4: x1 = W1 @ xn[k==0] + b1 (16 columns) ----
  if (wave < 4) {
    int mt  = wave;
    int col = (lane & 15) * Kk;      // columns with k==0
    v8f acc = {};
    acc = wmma_f16(loadA(W1, mt,  0, lane), loadB(sXN, col,  0, lane), acc);
    acc = wmma_f16(loadA(W1, mt, 32, lane), loadB(sXN, col, 32, lane), acc);
    int ob = mt*16 + ((lane >> 4) << 3);
    int nl = lane & 15;
    #pragma unroll
    for (int i = 0; i < 8; ++i) sX1[(ob+i)*TN + nl] = acc[i] + b1[ob+i];
  }
  __syncthreads();

  // ---- Phase 5: fused x2 = W2@xn, x3 = W3@xn, ptf = Wp2@pact ----
  {
    const int mt = wave & 3;
    const v16h aw2lo = loadA(W2,  mt,  0, lane), aw2hi = loadA(W2,  mt, 32, lane);
    const v16h aw3lo = loadA(W3,  mt,  0, lane), aw3hi = loadA(W3,  mt, 32, lane);
    const v16h awplo = loadA(Wp2, mt,  0, lane), awphi = loadA(Wp2, mt, 32, lane);
    const int ob = mt*16 + ((lane >> 4) << 3);
    float bias2[8], bias3[8];
    #pragma unroll
    for (int i = 0; i < 8; ++i) { bias2[i] = b2[ob+i]; bias3[i] = b3[ob+i]; }
    #pragma unroll
    for (int j = 0; j < 5; ++j) {
      int nt  = (wave + j*8) >> 2;
      int col = nt*16 + (lane & 15);
      v16h bx0 = loadB(sXN, col,  0, lane), bx1 = loadB(sXN, col, 32, lane);
      v16h bp0 = loadB(sA,  col,  0, lane), bp1 = loadB(sA,  col, 32, lane);
      v8f a2 = {}, a3 = {}, aP = {};
      a2 = wmma_f16(aw2lo, bx0, a2);  a2 = wmma_f16(aw2hi, bx1, a2);
      a3 = wmma_f16(aw3lo, bx0, a3);  a3 = wmma_f16(aw3hi, bx1, a3);
      aP = wmma_f16(awplo, bp0, aP);  aP = wmma_f16(awphi, bp1, aP);
      int nl = col / Kk;
      v8f r3, rf;
      #pragma unroll
      for (int i = 0; i < 8; ++i) {
        float pf = aP[i];
        r3[i] = a3[i] + bias3[i] + pf;                         // x3 + ptf
        rf[i] = sX1[(ob+i)*TN + nl] - (a2[i] + bias2[i]) + pf; // x1 - x2 + ptf
      }
      store8h(&sX3[col*ASTR + ob], r3);
      store8h(&sXF[col*ASTR + ob], rf);
    }
  }
  __syncthreads();

  // ---- Phase 6a: h = relu(Ww1 @ xfs)  (v8h-vectorized 640-MAC dots) ----
  if (tid < 128) {
    int o8 = tid >> 4, n = tid & 15;
    float acc = 0.f;
    #pragma unroll
    for (int k = 0; k < Kk; ++k) {
      const h16* xf = &sXF[(n*Kk + k)*ASTR];
      const h16* wt = &sW1t[(o8*Kk + k)*CH];
      #pragma unroll
      for (int m8 = 0; m8 < 8; ++m8) {
        v8h x = *(const v8h*)(xf + m8*8);
        v8h w = *(const v8h*)(wt + m8*8);
        #pragma unroll
        for (int i = 0; i < 8; ++i) acc += (float)w[i] * (float)x[i];
      }
    }
    sH[o8*TN + n] = acc > 0.f ? acc : 0.f;
  }
  __syncthreads();

  // ---- Phase 6b: wv = Ww2 @ h + bw2, softmax over k ----
  if (tid < 128) {
    int s = tid >> 4, n = tid & 15;
    float hv[8];
    #pragma unroll
    for (int j = 0; j < 8; ++j) hv[j] = sH[j*TN + n];
    float wvv[Kk];
    float mx = -1e30f;
    #pragma unroll
    for (int k = 0; k < Kk; ++k) {
      int so = s*Kk + k;
      float a = bw2[so];
      #pragma unroll
      for (int j = 0; j < 8; ++j) a += Ww2[so*8 + j] * hv[j];
      wvv[k] = a; mx = a > mx ? a : mx;
    }
    float sum = 0.f;
    #pragma unroll
    for (int k = 0; k < Kk; ++k) { wvv[k] = __expf(wvv[k] - mx); sum += wvv[k]; }
    float inv = 1.f / sum;
    #pragma unroll
    for (int k = 0; k < Kk; ++k) sWn[(s*TN + n)*Kk + k] = wvv[k] * inv;
  }
  __syncthreads();

  // ---- Phase 7: out1 = relu(sum_k w[o%8,n,k] * x3[o,n,k]) ----
  for (int idx = tid; idx < CH*TN; idx += 256) {
    int o = idx & 63, n = idx >> 6;
    const float* w = &sWn[((o & 7)*TN + n)*Kk];
    float acc = 0.f;
    #pragma unroll
    for (int k = 0; k < Kk; ++k) acc += w[k] * (float)sX3[(n*Kk + k)*ASTR + o];
    sO1[n*ASTR + o] = (h16)(acc > 0.f ? acc : 0.f);
  }
  __syncthreads();

  // ---- Phase 8: out = Wout @ out1 + bout + x ; write f32 ----
  if (wave < 4) {
    int mt = wave;
    int nl = lane & 15;
    int kh = (lane >> 4) << 4;
    v8f acc = {};
    acc = wmma_f16(loadA(Wout, mt,  0, lane), *(const v16h*)(sO1 + nl*ASTR +  0 + kh), acc);
    acc = wmma_f16(loadA(Wout, mt, 32, lane), *(const v16h*)(sO1 + nl*ASTR + 32 + kh), acc);
    int ob = mt*16 + ((lane >> 4) << 3);
    #pragma unroll
    for (int i = 0; i < 8; ++i) {
      int o = ob + i;
      float v = acc[i] + bout[o] + (float)sXN[(nl*Kk)*ASTR + o];  // + x (xn at k==0)
      out[((size_t)(b*CH + o))*Nn + n0 + nl] = v;
    }
  }
}

extern "C" void kernel_launch(void* const* d_in, const int* in_sizes, int n_in,
                              void* d_out, int out_size, void* d_ws, size_t ws_size,
                              hipStream_t stream) {
  const float* feats = (const float*)d_in[0];
  const float* cent  = (const float*)d_in[1];
  const float* smp   = (const float*)d_in[2];
  const float* W0    = (const float*)d_in[3];
  const float* b0    = (const float*)d_in[4];
  const float* W1    = (const float*)d_in[5];
  const float* b1    = (const float*)d_in[6];
  const float* W2    = (const float*)d_in[7];
  const float* b2    = (const float*)d_in[8];
  const float* W3    = (const float*)d_in[9];
  const float* b3    = (const float*)d_in[10];
  const float* Wp1   = (const float*)d_in[11];
  const float* Wp2   = (const float*)d_in[12];
  const float* Ww1   = (const float*)d_in[13];
  const float* Ww2   = (const float*)d_in[14];
  const float* bw2   = (const float*)d_in[15];
  const float* Wout  = (const float*)d_in[16];
  const float* bout  = (const float*)d_in[17];
  float* out = (float*)d_out;

  dim3 grid(Nn / TN, Bb);   // 512 x 8 workgroups
  dim3 block(256);          // 8 wave32
  pt_fused_kernel<<<grid, block, 0, stream>>>(feats, cent, smp,
      W0, b0, W1, b1, W2, b2, W3, b3, Wp1, Wp2, Ww1, Ww2, bw2, Wout, bout, out);
}
